// Model_hydro_lstm_regional_25683904430873
// MI455X (gfx1250) — matrix operand order repaired and checked by reading, not compiled, not run on hardware
//
#include <hip/hip_runtime.h>

// ---------------------------------------------------------------------------
// Problem constants (from the reference):
//   x: (32768, 758) f32, use first 730 cols
//   W_in: (4, 730), W_rec: (4,1), bias: (4,1), reg_w: (1,1), reg_b: (1,)
//   outputs (flat concat): q_t (32768), h_n (32769), c_n (32769)
// ---------------------------------------------------------------------------
#define T_STEPS   32768
#define F_STRIDE  758
#define K_DIM     730
#define K_PAD     736          // 23 * 32, zero-padded in LDS for weights
#define N_CHUNKS  23           // K chunks of 32
#define WAVES_PER_BLOCK 8      // blockDim = 256 = 8 wave32
#define TILES_M   (T_STEPS / 16)   // 2048

typedef __attribute__((ext_vector_type(16))) __bf16 v16bf;
typedef __attribute__((ext_vector_type(8)))  float  v8f;

// ---------------------------------------------------------------------------
// Kernel 1: xp[t][g] = sum_k x[t][k] * W[g][k]   via v_wmma_f32_16x16x32_bf16
// One wave computes a 16x16 tile (16 timesteps x 16 cols, only cols 0..3 kept).
// Bandwidth-bound: streams 99 MB of x once.
// ---------------------------------------------------------------------------
__global__ __launch_bounds__(256)
void xp_gemm_wmma(const float* __restrict__ x,
                  const float* __restrict__ w,      // (4, 730) row-major
                  float* __restrict__ xp)           // (32768, 4) row-major
{
    __shared__ __bf16 wlds[4 * K_PAD];              // 5888 bytes, zero-padded

    const int tid = threadIdx.x;

    // Cooperative load of weights -> LDS, bf16, zero pad K in [730, 736)
    for (int i = tid; i < 4 * K_PAD; i += blockDim.x) {
        const int g = i / K_PAD;
        const int k = i - g * K_PAD;
        const float v = (k < K_DIM) ? w[g * K_DIM + k] : 0.0f;
        wlds[i] = (__bf16)v;
    }
    __syncthreads();

    const int wave = tid >> 5;
    const int lane = tid & 31;
    const int hi   = lane >> 4;       // 0: lanes 0-15, 1: lanes 16-31
    const int mrow = lane & 15;       // A-matrix row for this lane
    const int n    = lane & 15;       // B/C/D column for this lane
    const int g    = n & 3;           // gate index (cols >=4 are discarded)

    const int t0 = (blockIdx.x * WAVES_PER_BLOCK + wave) * 16;
    const float* __restrict__ xrow = x + (size_t)(t0 + mrow) * F_STRIDE;
    const __bf16* __restrict__ wrow = &wlds[g * K_PAD];

    v8f acc = {};

    for (int c = 0; c < N_CHUNKS; ++c) {
        const int kbase = c * 32;

        // A (16x32 bf16): a[j] = K = kbase + hi*8 + j       (j in 0..7)
        //                 a[8+j] = K = kbase + 16 + hi*8 + j
        v16bf a;
        const int ka0 = kbase + hi * 8;
        const int ka1 = kbase + 16 + hi * 8;
        #pragma unroll
        for (int j = 0; j < 8; ++j) {
            a[j]     = (__bf16)xrow[ka0 + j];
            a[8 + j] = (__bf16)xrow[ka1 + j];
        }

        // B (32x16 bf16): b[j] = W[g][kbase + hi*8 + j], b[8+j] = W[g][kbase + 16 + hi*8 + j]
        v16bf b;
        #pragma unroll
        for (int j = 0; j < 8; ++j) {
            b[j]     = wrow[ka0 + j];
            b[8 + j] = wrow[ka1 + j];
        }

        // D = A * B + C  (f32 accumulate)
        acc = __builtin_amdgcn_wmma_f32_16x16x32_bf16(
            /*neg_a=*/false, a, /*neg_b=*/false, b,
            /*c_mod=*/(short)0, acc, /*reuse_a=*/false, /*reuse_b=*/false);
    }

    // Store: lane holds D rows (t0 + hi*8 + v), column n; keep only n < 4.
    if (n < 4) {
        #pragma unroll
        for (int v = 0; v < 8; ++v) {
            const int row = t0 + hi * 8 + v;
            xp[(size_t)row * 4 + n] = acc[v];
        }
    }
}

// ---------------------------------------------------------------------------
// Kernel 2: serial LSTM scan (scalar hidden state -> inherently sequential).
// Single thread walks all 32768 steps; writes q_t, h_n, c_n concatenated.
// ---------------------------------------------------------------------------
__global__ void lstm_scan(const float* __restrict__ xp,    // (32768, 4)
                          const float* __restrict__ wr4,   // (4,)
                          const float* __restrict__ b4,    // (4,)
                          const float* __restrict__ regw,  // (1,)
                          const float* __restrict__ regb,  // (1,)
                          const float* __restrict__ h0,
                          const float* __restrict__ c0,
                          float* __restrict__ out)
{
    if (threadIdx.x != 0 || blockIdx.x != 0) return;

    const float wr0 = wr4[0], wr1 = wr4[1], wr2 = wr4[2], wr3 = wr4[3];
    const float bb0 = b4[0],  bb1 = b4[1],  bb2 = b4[2],  bb3 = b4[3];
    const float rw = regw[0], rb = regb[0];

    float h = h0[0];
    float c = c0[0];

    float* __restrict__ q  = out;                       // 32768
    float* __restrict__ hn = out + T_STEPS;             // 32769
    float* __restrict__ cn = out + T_STEPS + T_STEPS + 1; // 32769

    hn[0] = h;
    cn[0] = c;

    for (int t = 0; t < T_STEPS; ++t) {
        const float* gv = xp + (size_t)t * 4;
        const float g0 = bb0 + wr0 * h + gv[0];
        const float g1 = bb1 + wr1 * h + gv[1];
        const float g2 = bb2 + wr2 * h + gv[2];
        const float g3 = bb3 + wr3 * h + gv[3];

        const float f  = 1.0f / (1.0f + expf(-g0));
        const float i  = 1.0f / (1.0f + expf(-g1));
        const float o  = 1.0f / (1.0f + expf(-g2));
        const float gg = tanhf(g3);

        c = f * c + i * gg;
        h = o * tanhf(c);

        q[t]      = h * rw + rb;
        hn[t + 1] = h;
        cn[t + 1] = c;
    }
}

// ---------------------------------------------------------------------------
// Launch
// ---------------------------------------------------------------------------
extern "C" void kernel_launch(void* const* d_in, const int* in_sizes, int n_in,
                              void* d_out, int out_size, void* d_ws, size_t ws_size,
                              hipStream_t stream) {
    const float* x    = (const float*)d_in[0];
    const float* w_in = (const float*)d_in[1];
    const float* w_rc = (const float*)d_in[2];
    const float* bias = (const float*)d_in[3];
    const float* regw = (const float*)d_in[4];
    const float* regb = (const float*)d_in[5];
    const float* h0   = (const float*)d_in[6];
    const float* c0   = (const float*)d_in[7];

    float* xp  = (float*)d_ws;            // 32768*4 floats = 512 KB scratch
    float* out = (float*)d_out;

    const int blocks = TILES_M / WAVES_PER_BLOCK;   // 256
    xp_gemm_wmma<<<blocks, 256, 0, stream>>>(x, w_in, xp);
    lstm_scan<<<1, 1, 0, stream>>>(xp, w_rc, bias, regw, regb, h0, c0, out);
}
